// SA_Attn_Mem_20495583937283
// MI455X (gfx1250) — matrix operand order repaired and checked by reading, not compiled, not run on hardware
//
#include <hip/hip_runtime.h>
#include <hip/hip_bf16.h>
#include <math.h>

// ---------------------------------------------------------------------------
// Types for CDNA5 WMMA (wave32): D(f32 16x16) = A(bf16 16x32) x B(bf16 32x16) + C
// ---------------------------------------------------------------------------
typedef __attribute__((ext_vector_type(16))) __bf16 bf16x16;
typedef __attribute__((ext_vector_type(8)))  __bf16 bf16x8;
typedef __attribute__((ext_vector_type(8)))  float  f32x8;
typedef __attribute__((ext_vector_type(4)))  float  f32x4;

__device__ __forceinline__ f32x8 wmma_bf16(bf16x16 a, bf16x16 b, f32x8 c) {
  return __builtin_amdgcn_wmma_f32_16x16x32_bf16(false, a, false, b, (short)0, c,
                                                 false, false);
}

// ---------------------------------------------------------------------------
// Problem constants
// ---------------------------------------------------------------------------
#define BB   4
#define CC   64
#define NN   4096           // H*W
#define NT16 256            // NN/16
#define BCN  1048576        // B*C*N  (f32 elements per output tensor)

// Workspace (bf16 elements).
#define WSZ    1048576ull
#define QH_OFF (0ull*WSZ)   // Q_h  A-packed  [b][qt(256)][kc(2)] x 512
#define KH_OFF (1ull*WSZ)   // K_h  B-packed  [b][rt(2)][nt(256)] x 512   rows=d
#define KM_OFF (2ull*WSZ)   // K_m  B-packed
#define VH_OFF (3ull*WSZ)   // V_h  B-packed-T [b][nchunk(128)][ct(4)] x 512 rows=key
#define VM_OFF (4ull*WSZ)   // V_m  B-packed-T
#define HP_OFF (5ull*WSZ)   // h    B-packed  rows=c
#define ZH_OFF (6ull*WSZ)   // Z_h  B-packed  rows=c
#define ZM_OFF (7ull*WSZ)   // Z_m  B-packed  rows=c
// One-time bf16 A-packed weights (k0 writes, k1/k3 read):
#define W5_OFF (8ull*WSZ)                 // 5 x [4 dt][2 kc] tiles = 5*4096
#define WZ_OFF (W5_OFF + 5ull*4096ull)    // Wz: [8 dt][4 kc] = 32 tiles
#define WM_OFF (WZ_OFF + 32ull*512ull)    // Wm: [12 dt][6 kc] = 72 tiles

// ===========================================================================
// Kernel 0: convert all weight matrices to A-packed bf16 tiles (one time).
// A-pack position of element (r,k): tile (r>>4, k>>5);
//   lane' = (r&15) + 16*((k&31)>=16), slot' = k&15
// grid = 7 blocks (one per matrix), 256 threads.
// ===========================================================================
__device__ __forceinline__ void packA_mat(const float* __restrict__ W, int R,
                                          int K, __bf16* __restrict__ dst, int t) {
  const int ntk = K >> 5;                      // k-chunk tiles per row band
  for (int e = t * 4; e < R * K; e += 256 * 4) {
    const int r = e / K, k = e % K;            // 4 consecutive k, same row
    f32x4 x = *(const f32x4*)(W + e);
    #pragma unroll
    for (int j = 0; j < 4; ++j) {
      const int kk = k + j;
      const int tile = (r >> 4) * ntk + (kk >> 5);
      const int kl = kk & 31;
      const int lp = (r & 15) + ((kl >= 16) ? 16 : 0);
      const int sp = kl & 15;
      dst[tile * 512 + lp * 16 + sp] = (__bf16)x[j];
    }
  }
}

__global__ __launch_bounds__(256) void k0_packw(
    const float* __restrict__ Wq, const float* __restrict__ Wk,
    const float* __restrict__ Wv, const float* __restrict__ Wk2,
    const float* __restrict__ Wv2, const float* __restrict__ Wz,
    const float* __restrict__ Wm, __bf16* __restrict__ ws)
{
  const int which = blockIdx.x;
  const int t = threadIdx.x;
  switch (which) {
    case 0: packA_mat(Wq,  64, 64,  ws + W5_OFF + 0 * 4096, t); break;
    case 1: packA_mat(Wk,  64, 64,  ws + W5_OFF + 1 * 4096, t); break;
    case 2: packA_mat(Wv,  64, 64,  ws + W5_OFF + 2 * 4096, t); break;
    case 3: packA_mat(Wk2, 64, 64,  ws + W5_OFF + 3 * 4096, t); break;
    case 4: packA_mat(Wv2, 64, 64,  ws + W5_OFF + 4 * 4096, t); break;
    case 5: packA_mat(Wz,  128, 128, ws + WZ_OFF, t); break;
    default:packA_mat(Wm,  192, 192, ws + WM_OFF, t); break;
  }
}

// ---------------------------------------------------------------------------
// Kernel 1 helper: one projection (Wa A-packed bf16 @ slab + bias), PATH picks
// the compile-time pack layout of the result:
//   PATH 0: Q  -> A-packed (row = n, K = d); per-lane 8 consecutive bf16
//   PATH 1: K  -> B-packed (rows = d);      per-lane 8 consecutive bf16
//   PATH 2: V  -> B-packed-T (rows = key);  per-lane 8 stores, stride 32B
// ---------------------------------------------------------------------------
template <int PATH>
__device__ __forceinline__ void k1_job(
    const __bf16* __restrict__ Wa, const float* __restrict__ bias,
    const __bf16* __restrict__ slab, __bf16* __restrict__ ws, size_t baseOff,
    int b, int n0, int wave, int lane, int hi)
{
  #pragma unroll
  for (int tt = 0; tt < 2; ++tt) {
    const int tile = wave * 2 + tt;          // 0..15
    const int ti = tile >> 2, tj = tile & 3; // row-tile, col-tile
    f32x8 acc = *(const f32x8*)(bias + ti * 16 + hi);   // 8 consecutive rows
    #pragma unroll
    for (int kc = 0; kc < 2; ++kc) {
      bf16x16 a  = *(const bf16x16*)(Wa + (ti * 2 + kc) * 512 + lane * 16);
      bf16x16 bb = *(const bf16x16*)(slab + (kc * 4 + tj) * 512 + lane * 16);
      acc = wmma_bf16(a, bb, acc);
    }
    if (PATH == 0) {
      // d = ti*16 + hi + v  ->  kc = ti>>1, lane half = ti&1, slot = hi + v: contiguous in v
      const int qt = (n0 >> 4) + tj;
      const int lp = (lane & 15) + ((ti & 1) ? 16 : 0);
      bf16x8 r;
      #pragma unroll
      for (int v = 0; v < 8; ++v) r[v] = (__bf16)acc[v];
      *(bf16x8*)(ws + baseOff +
                 (((size_t)(b * NT16 + qt)) * 2 + (ti >> 1)) * 512 +
                 lp * 16 + hi) = r;
    } else if (PATH == 1) {
      // slot = hi + v: contiguous in v
      const int nt = (n0 >> 4) + tj;
      const int lp = (lane & 15) + ((ti & 1) ? 16 : 0);
      bf16x8 r;
      #pragma unroll
      for (int v = 0; v < 8; ++v) r[v] = (__bf16)acc[v];
      *(bf16x8*)(ws + baseOff +
                 (((size_t)(b * 2 + (ti >> 1))) * NT16 + nt) * 512 +
                 lp * 16 + hi) = r;
    } else {
      // transposed pack: lane' = hi + v + 16*(tj&1), slot' = lane&15
      const int chunk = (n0 >> 5) + (tj >> 1);
      __bf16* dst = ws + baseOff +
                    (((size_t)(b * 128 + chunk)) * 4 + ti) * 512 +
                    (hi + ((tj & 1) ? 16 : 0)) * 16 + (lane & 15);
      #pragma unroll
      for (int v = 0; v < 8; ++v) dst[v * 16] = (__bf16)acc[v];
    }
  }
}

// ===========================================================================
// Kernel 1: projections Q_h, K_h, V_h, K_m, V_m  (+ pack h to bf16)
// grid (64 slabs, B), 256 threads (8 waves). Each block: one 64-column slab.
// ===========================================================================
__global__ __launch_bounds__(256) void k1_proj(
    const float* __restrict__ h,  const float* __restrict__ mm,
    const float* __restrict__ bq, const float* __restrict__ bk,
    const float* __restrict__ bk2,const float* __restrict__ bv,
    const float* __restrict__ bv2, __bf16* __restrict__ ws)
{
  __shared__ __align__(32) __bf16 slabH[4096];   // B-packed [tile(8)][512]
  __shared__ __align__(32) __bf16 slabM[4096];
  const int t  = threadIdx.x;
  const int n0 = blockIdx.x * 64;
  const int b  = blockIdx.y;

  // ---- load f32 slabs, convert, scatter into B-packed LDS ----
  #pragma unroll
  for (int p = 0; p < 4; ++p) {
    const int c  = p * 16 + (t >> 4);
    const int nl = (t & 15) * 4;
    const size_t gsrc = ((size_t)(b * CC + c)) * NN + n0 + nl;
    f32x4 xh = *(const f32x4*)(h  + gsrc);
    f32x4 xm = *(const f32x4*)(mm + gsrc);
    #pragma unroll
    for (int j = 0; j < 4; ++j) {
      const int n    = nl + j;
      const int tile = (c >> 5) * 4 + (n >> 4);
      const int idx  = tile * 512 +
                       ((n & 15) + (((c & 31) >= 16) ? 16 : 0)) * 16 + (c & 15);
      slabH[idx] = (__bf16)xh[j];
      slabM[idx] = (__bf16)xm[j];
    }
  }
  __syncthreads();

  const int wave = t >> 5, lane = t & 31;
  const int hi = (lane & 16) ? 8 : 0;

  // h -> global bf16 B-packed: bulk contiguous copy (LDS tile == global tile)
  {
    const int tile = t >> 5;            // 0..7
    const int off  = (t & 31) * 16;     // 16 elements per thread
    const int rt = tile >> 2, nt = tile & 3;
    bf16x16 v16 = *(const bf16x16*)(slabH + tile * 512 + off);
    *(bf16x16*)(ws + HP_OFF +
                (((size_t)(b * 2 + rt)) * NT16 + (n0 >> 4) + nt) * 512 + off) = v16;
  }

  // five projections; weights pre-packed bf16, pack path compile-time
  const __bf16* w5 = ws + W5_OFF;
  k1_job<0>(w5 + 0 * 4096, bq,  slabH, ws, QH_OFF, b, n0, wave, lane, hi);
  k1_job<1>(w5 + 1 * 4096, bk,  slabH, ws, KH_OFF, b, n0, wave, lane, hi);
  k1_job<2>(w5 + 2 * 4096, bv,  slabH, ws, VH_OFF, b, n0, wave, lane, hi);
  k1_job<1>(w5 + 3 * 4096, bk2, slabM, ws, KM_OFF, b, n0, wave, lane, hi);
  k1_job<2>(w5 + 4 * 4096, bv2, slabM, ws, VM_OFF, b, n0, wave, lane, hi);
}

// ===========================================================================
// Kernel 2: flash attention, one 16-query tile per block, one stream (h|m).
// grid (256 q-tiles, B, 2 streams), 128 threads = 4 waves, each wave owns
// 1024 keys with private online-softmax state; merged through LDS at the end.
// ===========================================================================
__global__ __launch_bounds__(128) void k2_attn(__bf16* __restrict__ ws)
{
  __shared__ __align__(32) __bf16 Pst[4][512];   // per-wave P (D->A relayout)
  __shared__ float sM[64], sL[64];
  __shared__ float sAcc[64 * 64];                // [wave*16+q][c]

  const int t = threadIdx.x, wave = t >> 5, lane = t & 31;
  const int qt = blockIdx.x, b = blockIdx.y, strm = blockIdx.z;
  const int hi = (lane & 16) ? 8 : 0;

  const __bf16* Kp = ws + (strm ? KM_OFF : KH_OFF);
  const __bf16* Vp = ws + (strm ? VM_OFF : VH_OFF);
  __bf16*       Zp = ws + (strm ? ZM_OFF : ZH_OFF);

  bf16x16 qa[2];
  #pragma unroll
  for (int kc = 0; kc < 2; ++kc)
    qa[kc] = *(const bf16x16*)(ws + QH_OFF +
                (((size_t)(b * NT16 + qt)) * 2 + kc) * 512 + lane * 16);

  float mI[8], lI[8]; f32x8 acc[4];
  #pragma unroll
  for (int v = 0; v < 8; ++v) { mI[v] = -__builtin_inff(); lI[v] = 0.f; }
  #pragma unroll
  for (int ct = 0; ct < 4; ++ct)
    #pragma unroll
    for (int v = 0; v < 8; ++v) acc[ct][v] = 0.f;

  __bf16* Pw = &Pst[wave][0];

  for (int it = 0; it < 32; ++it) {
    const int key0 = wave * 1024 + it * 32;
    f32x8 S0, S1;
    #pragma unroll
    for (int v = 0; v < 8; ++v) { S0[v] = 0.f; S1[v] = 0.f; }
    #pragma unroll
    for (int hc = 0; hc < 2; ++hc) {        // K-dim = HD in two 32-chunks
      const __bf16* kb = Kp + (((size_t)(b * 2 + hc)) * NT16 + (key0 >> 4)) * 512;
      bf16x16 k0 = *(const bf16x16*)(kb + lane * 16);
      bf16x16 k1 = *(const bf16x16*)(kb + 512 + lane * 16);
      S0 = wmma_bf16(qa[hc], k0, S0);
      S1 = wmma_bf16(qa[hc], k1, S1);
    }
    // ---- online softmax over 32 keys; row stats reduced across 16 lanes ----
    #pragma unroll
    for (int v = 0; v < 8; ++v) {
      float rm = fmaxf(S0[v], S1[v]);
      rm = fmaxf(rm, __shfl_xor(rm, 1, 32));
      rm = fmaxf(rm, __shfl_xor(rm, 2, 32));
      rm = fmaxf(rm, __shfl_xor(rm, 4, 32));
      rm = fmaxf(rm, __shfl_xor(rm, 8, 32));
      const float mn = fmaxf(mI[v], rm);
      const float sc = __expf(mI[v] - mn);
      mI[v] = mn;
      const float p0 = __expf(S0[v] - mn);
      const float p1 = __expf(S1[v] - mn);
      float rs = p0 + p1;
      rs += __shfl_xor(rs, 1, 32); rs += __shfl_xor(rs, 2, 32);
      rs += __shfl_xor(rs, 4, 32); rs += __shfl_xor(rs, 8, 32);
      lI[v] = lI[v] * sc + rs;
      // D-layout -> A-packed LDS: (q=v+hi, key = kt*16 + lane&15)
      const int lp = (v + hi);
      Pw[lp * 16 + (lane & 15)       ] = (__bf16)p0;   // kt=0
      Pw[(lp + 16) * 16 + (lane & 15)] = (__bf16)p1;   // kt=1
      acc[0][v] *= sc; acc[1][v] *= sc; acc[2][v] *= sc; acc[3][v] *= sc;
    }
    asm volatile("s_wait_dscnt 0" ::: "memory");   // wave-local LDS RAW fence
    bf16x16 pa = *(const bf16x16*)(Pw + lane * 16);
    const __bf16* vb = Vp + (((size_t)(b * 128 + (key0 >> 5))) * 4) * 512;
    #pragma unroll
    for (int ct = 0; ct < 4; ++ct) {
      bf16x16 vv = *(const bf16x16*)(vb + ct * 512 + lane * 16);
      acc[ct] = wmma_bf16(pa, vv, acc[ct]);
    }
  }

  // ---- merge 4 waves ----
  if ((lane & 15) == 0) {
    #pragma unroll
    for (int v = 0; v < 8; ++v) { sM[wave * 16 + v + hi] = mI[v];
                                  sL[wave * 16 + v + hi] = lI[v]; }
  }
  #pragma unroll
  for (int ct = 0; ct < 4; ++ct)
    #pragma unroll
    for (int v = 0; v < 8; ++v)
      sAcc[(wave * 16 + v + hi) * 64 + ct * 16 + (lane & 15)] = acc[ct][v];
  __syncthreads();

  for (int e = t; e < 1024; e += 128) {
    const int q = e >> 6, c = e & 63;
    float M = -__builtin_inff();
    for (int w = 0; w < 4; ++w) M = fmaxf(M, sM[w * 16 + q]);
    float L = 0.f, z = 0.f;
    for (int w = 0; w < 4; ++w) {
      const float f = __expf(sM[w * 16 + q] - M);
      L += f * sL[w * 16 + q];
      z += f * sAcc[(w * 16 + q) * 64 + c];
    }
    const float outv = z / L;
    const int n  = qt * 16 + q;
    const int rt = c >> 5;
    const int lp = (n & 15) + (((c & 31) >= 16) ? 16 : 0);
    const int sp = c & 15;
    Zp[(((size_t)(b * 2 + rt)) * NT16 + qt) * 512 + lp * 16 + sp] = (__bf16)outv;
  }
}

// ===========================================================================
// Kernel 3: Z2 = Wz @ [Z_h;Z_m] + bz ; Out = Wm @ [Z2;h] + bm ; gating.
// grid (64 slabs, B), 256 threads (8 waves). Each wave's GEMM2 jobs pair
// (mo,mg,mi) row tiles so gating stays entirely in registers.
// ===========================================================================
__global__ __launch_bounds__(256) void k3_out(
    const float* __restrict__ mm,
    const float* __restrict__ bz, const float* __restrict__ bm,
    const __bf16* __restrict__ ws, float* __restrict__ out)
{
  __shared__ __align__(32) __bf16 Z2[16 * 512];  // B-packed [rt(4)][nt(4)]
  const int t = threadIdx.x, wave = t >> 5, lane = t & 31;
  const int n0 = blockIdx.x * 64, b = blockIdx.y;
  const int hi = (lane & 16) ? 8 : 0;
  const __bf16* wzA = ws + WZ_OFF;
  const __bf16* wmA = ws + WM_OFF;

  // ---- GEMM1: Z2[128 x 64slab] ----
  {
    const int dt = wave;
    #pragma unroll
    for (int nt = 0; nt < 4; ++nt) {
      f32x8 acc = *(const f32x8*)(bz + dt * 16 + hi);
      #pragma unroll
      for (int kc = 0; kc < 4; ++kc) {
        bf16x16 a = *(const bf16x16*)(wzA + (dt * 4 + kc) * 512 + lane * 16);
        const size_t zoff = (kc < 2) ? ZH_OFF : ZM_OFF;
        const int rt = kc & 1;
        const __bf16* bp = ws + zoff +
            (((size_t)(b * 2 + rt)) * NT16 + (n0 >> 4) + nt) * 512;
        bf16x16 bb = *(const bf16x16*)(bp + lane * 16);
        acc = wmma_bf16(a, bb, acc);
      }
      // row = dt*16 + hi + v -> slot = hi + v: one contiguous b128 LDS store
      const int tilebase = (dt >> 1) * 4 + nt;
      const int lp = (lane & 15) + ((dt & 1) ? 16 : 0);
      bf16x8 r;
      #pragma unroll
      for (int v = 0; v < 8; ++v) r[v] = (__bf16)acc[v];
      *(bf16x8*)(&Z2[tilebase * 512 + lp * 16 + hi]) = r;
    }
  }
  __syncthreads();

  // ---- GEMM2 + gating ----
  #pragma unroll
  for (int jb = 0; jb < 2; ++jb) {
    const int job = wave * 2 + jb;
    const int g = job >> 2, nt = job & 3;
    f32x8 aO = *(const f32x8*)(bm + (g    ) * 16 + hi);
    f32x8 aG = *(const f32x8*)(bm + (g + 4) * 16 + hi);
    f32x8 aI = *(const f32x8*)(bm + (g + 8) * 16 + hi);
    #pragma unroll
    for (int kc = 0; kc < 6; ++kc) {
      bf16x16 bb;
      if (kc < 4) {
        bb = *(const bf16x16*)(&Z2[(kc * 4 + nt) * 512] + lane * 16);
      } else {
        const int rt = kc - 4;
        bb = *(const bf16x16*)(ws + HP_OFF +
              (((size_t)(b * 2 + rt)) * NT16 + (n0 >> 4) + nt) * 512 + lane * 16);
      }
      aO = wmma_bf16(*(const bf16x16*)(wmA + ((g    ) * 6 + kc) * 512 + lane * 16), bb, aO);
      aG = wmma_bf16(*(const bf16x16*)(wmA + ((g + 4) * 6 + kc) * 512 + lane * 16), bb, aG);
      aI = wmma_bf16(*(const bf16x16*)(wmA + ((g + 8) * 6 + kc) * 512 + lane * 16), bb, aI);
    }
    #pragma unroll
    for (int v = 0; v < 8; ++v) {
      const int c = g * 16 + v + hi;
      const int n = n0 + nt * 16 + (lane & 15);
      const size_t gi = ((size_t)(b * CC + c)) * NN + n;
      const float m_in = mm[gi];
      const float si = 1.f / (1.f + __expf(-aI[v]));
      const float nm = (1.f - si) * m_in + si * tanhf(aG[v]);
      const float nh = (1.f / (1.f + __expf(-aO[v]))) * nm;
      out[gi]       = nh;   // new_h
      out[BCN + gi] = nm;   // new_m
    }
  }
}

// ===========================================================================
extern "C" void kernel_launch(void* const* d_in, const int* in_sizes, int n_in,
                              void* d_out, int out_size, void* d_ws, size_t ws_size,
                              hipStream_t stream) {
  const float* h   = (const float*)d_in[0];
  const float* m   = (const float*)d_in[1];
  const float* Wq  = (const float*)d_in[2];
  const float* bq  = (const float*)d_in[3];
  const float* Wk  = (const float*)d_in[4];
  const float* bk  = (const float*)d_in[5];
  const float* Wk2 = (const float*)d_in[6];
  const float* bk2 = (const float*)d_in[7];
  const float* Wv  = (const float*)d_in[8];
  const float* bv  = (const float*)d_in[9];
  const float* Wv2 = (const float*)d_in[10];
  const float* bv2 = (const float*)d_in[11];
  const float* Wz  = (const float*)d_in[12];
  const float* bz  = (const float*)d_in[13];
  const float* Wm  = (const float*)d_in[14];
  const float* bm  = (const float*)d_in[15];
  __bf16* ws = (__bf16*)d_ws;
  float* out = (float*)d_out;

  k0_packw<<<dim3(7), 256, 0, stream>>>(Wq, Wk, Wv, Wk2, Wv2, Wz, Wm, ws);
  k1_proj<<<dim3(64, BB), 256, 0, stream>>>(h, m, bq, bk, bk2, bv, bv2, ws);
  k2_attn<<<dim3(NT16, BB, 2), 128, 0, stream>>>(ws);
  k3_out<<<dim3(64, BB), 256, 0, stream>>>(m, bz, bm, ws, out);
}
